// GINEClassifierFeatsNoEmb_66254165508833
// MI455X (gfx1250) — matrix operand order repaired and checked by reading, not compiled, not run on hardware
//
#include <hip/hip_runtime.h>
#include <hip/hip_bf16.h>
#include <stdint.h>

// ---------------- problem constants ----------------
#define NUM_TYPES 200
#define TOK_DIM 1024
#define SMALL_DIM 2
#define CH 1227
#define CHP 1280            // padded channels (multiple of 128)
#define NUM_LAYERS 4
#define NUM_EDGE_TYPES 3
#define N_NODES 60000
#define N_EDGES 120000
#define N_GRAPHS 64
#define BN_EPS 1e-5f
#define KTILES (CHP / 32)

typedef __bf16 bf16_t;
typedef __attribute__((ext_vector_type(16))) __bf16 v16bf;
typedef __attribute__((ext_vector_type(8)))  float  v8f;

// ---------------- CDNA5 async-to-LDS helpers ----------------
// GLOBAL_LOAD_ASYNC_TO_LDS_B128: VDST = LDS byte address (VGPR), VADDR = 64-bit
// global address. Tracked with ASYNCcnt (ISA 15.18.3 op 98).
__device__ __forceinline__ void async_copy16(uint32_t lds_addr, const void* gptr) {
  asm volatile("global_load_async_to_lds_b128 %0, %1, off"
               :: "v"(lds_addr), "v"(gptr)
               : "memory");
}

__device__ __forceinline__ void wait_async0() {
#if defined(__has_builtin) && __has_builtin(__builtin_amdgcn_s_wait_asynccnt)
  __builtin_amdgcn_s_wait_asynccnt(0);
#else
  asm volatile("s_wait_asynccnt 0x0" ::: "memory");
#endif
}

// ---------------- utility kernels ----------------
__global__ void k_zero_u32(uint32_t* __restrict__ p, long long n) {
  long long i = (long long)blockIdx.x * blockDim.x + threadIdx.x;
  long long stride = (long long)gridDim.x * blockDim.x;
  for (; i < n; i += stride) p[i] = 0u;
}

__global__ void k_build_features(const int* __restrict__ xt, const int* __restrict__ xk,
                                 const float* __restrict__ xs, bf16_t* __restrict__ h) {
  int i = blockIdx.x * blockDim.x + threadIdx.x;
  if (i >= N_NODES) return;
  bf16_t* row = h + (long long)i * CHP;
  int t = xt[i];
  if (t >= 0 && t < NUM_TYPES) row[t] = (bf16_t)1.0f;
  int tok = xk[i];
  tok = tok < 0 ? 0 : (tok > TOK_DIM ? TOK_DIM : tok);
  row[NUM_TYPES + tok] = (bf16_t)1.0f;
  row[NUM_TYPES + TOK_DIM + 1] = (bf16_t)xs[i * 2 + 0];
  row[NUM_TYPES + TOK_DIM + 2] = (bf16_t)xs[i * 2 + 1];
}

// Pack W1/W2 for all layers into transposed + padded bf16 layout:
// Wt[(l*2+m)][n*CHP + k] = W_m[l][k*CH + n]  (zeros in padding)
__global__ void k_pack_w(const float* __restrict__ W1, const float* __restrict__ W2,
                         bf16_t* __restrict__ Wt) {
  const long long per = (long long)CHP * CHP;
  const long long total = (long long)NUM_LAYERS * 2 * per;
  long long idx = (long long)blockIdx.x * blockDim.x + threadIdx.x;
  long long stride = (long long)gridDim.x * blockDim.x;
  for (; idx < total; idx += stride) {
    int lm = (int)(idx / per);
    long long r = idx % per;
    int n = (int)(r / CHP);
    int k = (int)(r % CHP);
    int l = lm >> 1, m = lm & 1;
    float v = 0.0f;
    if (n < CH && k < CH) {
      const float* W = (m == 0 ? W1 : W2) + (long long)l * CH * CH;
      v = W[(long long)k * CH + n];
    }
    Wt[idx] = (bf16_t)v;
  }
}

// GINE message: m = relu(h[src] + We[l][edge_type] + be[l]); scatter-add to aggr[dst]
__global__ void k_edge_scatter(const bf16_t* __restrict__ h, const int* __restrict__ ei,
                               const int* __restrict__ et, const float* __restrict__ We_l,
                               const float* __restrict__ be_l, float* __restrict__ aggr) {
  int e = blockIdx.x;
  int src = ei[e];
  int dst = ei[N_EDGES + e];
  const float* we = We_l + (long long)et[e] * CH;
  const bf16_t* hs = h + (long long)src * CHP;
  float* ad = aggr + (long long)dst * CHP;
  for (int c = threadIdx.x; c < CH; c += blockDim.x) {
    float m = (float)hs[c] + we[c] + be_l[c];
    if (m > 0.0f) atomicAdd(&ad[c], m);
  }
}

// zin = bf16(h + aggr)
__global__ void k_pack_zin(const bf16_t* __restrict__ h, const float* __restrict__ aggr,
                           bf16_t* __restrict__ zin) {
  long long total = (long long)N_NODES * CHP;
  long long i = (long long)blockIdx.x * blockDim.x + threadIdx.x;
  long long stride = (long long)gridDim.x * blockDim.x;
  for (; i < total; i += stride) zin[i] = (bf16_t)((float)h[i] + aggr[i]);
}

// ---------------- WMMA bf16 GEMM ----------------
// C[M x CHP] = act( A[M x CHP](bf16) @ Bt^T + bias ),  Bt stored N-major.
// Block tile 128x128, BK=32, 8 wave32s; each wave: 2x4 tiles of 16x16.
// Tiles staged with global_load_async_to_lds_b128, double-buffered.
// A-tile rows stored K-permuted [0..7 | 16..23 | 8..15 | 24..31] so every WMMA
// fragment is one contiguous 32-byte LDS load matching the ISA VGPR layout.
template <bool RELU, bool OUT_BF16>
__global__ __launch_bounds__(256) void k_gemm(const bf16_t* __restrict__ A,
                                              const bf16_t* __restrict__ Bt,
                                              const float* __restrict__ bias,
                                              bf16_t* __restrict__ outb,
                                              float* __restrict__ outf) {
  __shared__ __align__(32) bf16_t sA[2][128][32];
  __shared__ __align__(32) bf16_t sB[2][128][32];   // N-major: sB[b][n][k]

  const int tid  = threadIdx.x;
  const int wave = tid >> 5;
  const int lane = tid & 31;
  const int half = lane >> 4;
  const int lan  = lane & 15;
  const int waveM = wave & 3;      // 4 waves along M (4*32 = 128)
  const int waveN = wave >> 2;     // 2 waves along N (2*64 = 128)
  const long long tileM = (long long)blockIdx.x * 128;
  const int tileN = blockIdx.y * 128;

  v8f acc[2][4] = {};

  auto issue_tiles = [&](int k0, int b) {
#pragma unroll
    for (int i = 0; i < 2; i++) {
      int idx = tid + i * 256;                 // 0..511
      int r = idx >> 2;
      int s = idx & 3;                         // 8-element K segment
      int slot = ((s & 1) << 1) | (s >> 1);    // 0,1,2,3 -> 0,2,1,3 (A permute)
      long long gr = tileM + r;
      bf16_t* la = &sA[b][r][slot * 8];
      if (gr < N_NODES) {
        async_copy16((uint32_t)(uintptr_t)la, A + gr * CHP + k0 + s * 8);
      } else {
        *(uint4*)la = make_uint4(0u, 0u, 0u, 0u);
      }
      bf16_t* lb = &sB[b][r][slot * 8];        // B: same K permute as A
      async_copy16((uint32_t)(uintptr_t)lb,
                   Bt + (long long)(tileN + r) * CHP + k0 + s * 8);
    }
  };

  int buf = 0;
  issue_tiles(0, 0);

  for (int kt = 0; kt < KTILES; kt++) {
    wait_async0();        // my async writes for tile kt complete
    __syncthreads();      // everyone's writes visible; prior reads of buf^1 done
    if (kt + 1 < KTILES) issue_tiles((kt + 1) * 32, buf ^ 1);

    v16bf afr[2], bfr[4];
#pragma unroll
    for (int mt = 0; mt < 2; mt++)
      afr[mt] = *(const v16bf*)(&sA[buf][waveM * 32 + mt * 16 + lan][half * 16]);
#pragma unroll
    for (int nt = 0; nt < 4; nt++)
      bfr[nt] = *(const v16bf*)(&sB[buf][waveN * 64 + nt * 16 + lan][half * 16]);

#pragma unroll
    for (int mt = 0; mt < 2; mt++)
#pragma unroll
      for (int nt = 0; nt < 4; nt++)
        acc[mt][nt] = __builtin_amdgcn_wmma_f32_16x16x32_bf16(
            false, afr[mt], false, bfr[nt], (short)0, acc[mt][nt], false, false);

    buf ^= 1;
  }

  // epilogue: D layout — lanes 0-15 cols, VGPR v -> row v (+8 for lanes 16-31)
  for (int mt = 0; mt < 2; mt++)
    for (int nt = 0; nt < 4; nt++) {
      int col = tileN + waveN * 64 + nt * 16 + lan;
      float bv = (col < CH) ? bias[col] : 0.0f;
#pragma unroll
      for (int v = 0; v < 8; v++) {
        long long grow = tileM + waveM * 32 + mt * 16 + v + half * 8;
        if (grow < N_NODES) {
          float x = acc[mt][nt][v] + bv;
          if (RELU && x < 0.0f) x = 0.0f;
          if (OUT_BF16) outb[grow * CHP + col] = (bf16_t)x;
          else          outf[grow * CHP + col] = x;
        }
      }
    }
}

// ---------------- batchnorm ----------------
__global__ void k_bn_stats(const float* __restrict__ z, float* __restrict__ stats) {
  int c = blockIdx.x * 256 + threadIdx.x;           // c < CHP
  const long long chunk = (N_NODES + 63) / 64;
  long long r0 = (long long)blockIdx.y * chunk;
  long long r1 = r0 + chunk; if (r1 > N_NODES) r1 = N_NODES;
  float s = 0.0f, s2 = 0.0f;
  for (long long r = r0; r < r1; r++) {
    float v = z[r * CHP + c];
    s += v; s2 += v * v;
  }
  atomicAdd(&stats[c], s);
  atomicAdd(&stats[CHP + c], s2);
}

__global__ void k_bn_finalize(float* __restrict__ stats) {
  int c = blockIdx.x * blockDim.x + threadIdx.x;
  if (c >= CHP) return;
  float mean = stats[c] * (1.0f / N_NODES);
  float var  = stats[CHP + c] * (1.0f / N_NODES) - mean * mean;
  if (var < 0.0f) var = 0.0f;
  stats[c] = mean;
  stats[CHP + c] = rsqrtf(var + BN_EPS);
}

__global__ void k_bn_apply(const float* __restrict__ z, const float* __restrict__ stats,
                           const float* __restrict__ gamma, const float* __restrict__ beta,
                           bf16_t* __restrict__ h) {
  long long total = (long long)N_NODES * CHP;
  long long i = (long long)blockIdx.x * blockDim.x + threadIdx.x;
  long long stride = (long long)gridDim.x * blockDim.x;
  for (; i < total; i += stride) {
    int c = (int)(i % CHP);
    float o = 0.0f;
    if (c < CH) {
      float v = (z[i] - stats[c]) * stats[CHP + c] * gamma[c] + beta[c];
      o = v > 0.0f ? v : 0.0f;
    }
    h[i] = (bf16_t)o;
  }
}

// ---------------- pooling + head ----------------
__global__ void k_pool(const bf16_t* __restrict__ h, const int* __restrict__ batch,
                       float* __restrict__ sums, float* __restrict__ cnts) {
  int i = blockIdx.x;
  int b = batch[i];
  const bf16_t* row = h + (long long)i * CHP;
  float* s = sums + (long long)b * CHP;
  for (int c = threadIdx.x; c < CH; c += blockDim.x) {
    float v = (float)row[c];
    if (v != 0.0f) atomicAdd(&s[c], v);
  }
  if (threadIdx.x == 0) atomicAdd(&cnts[b], 1.0f);
}

__global__ __launch_bounds__(256) void k_head(const float* __restrict__ sums,
                                              const float* __restrict__ cnts,
                                              const float* __restrict__ Wh1,
                                              const float* __restrict__ bh1,
                                              const float* __restrict__ Wh2,
                                              const float* __restrict__ bh2,
                                              float* __restrict__ out) {
  __shared__ float hg[CH];
  __shared__ float t1[CH];
  int g = blockIdx.x;
  float inv = 1.0f / fmaxf(cnts[g], 1.0f);
  for (int c = threadIdx.x; c < CH; c += 256) hg[c] = sums[(long long)g * CHP + c] * inv;
  __syncthreads();
  for (int oc = threadIdx.x; oc < CH; oc += 256) {
    float a = bh1[oc];
    for (int k = 0; k < CH; k++) a += hg[k] * Wh1[(long long)k * CH + oc];
    t1[oc] = a > 0.0f ? a : 0.0f;
  }
  __syncthreads();
  if (threadIdx.x < 2) {
    float a = bh2[threadIdx.x];
    for (int k = 0; k < CH; k++) a += t1[k] * Wh2[k * 2 + threadIdx.x];
    out[g * 2 + threadIdx.x] = a;
  }
}

// ---------------- orchestration ----------------
extern "C" void kernel_launch(void* const* d_in, const int* in_sizes, int n_in,
                              void* d_out, int out_size, void* d_ws, size_t ws_size,
                              hipStream_t stream) {
  (void)in_sizes; (void)n_in; (void)out_size; (void)ws_size;
  const int*   x_type   = (const int*)d_in[0];
  const int*   x_tok    = (const int*)d_in[1];
  const float* x_small  = (const float*)d_in[2];
  const int*   edge_idx = (const int*)d_in[3];
  const int*   edge_typ = (const int*)d_in[4];
  const int*   batch    = (const int*)d_in[5];
  const float* We  = (const float*)d_in[6];
  const float* be  = (const float*)d_in[7];
  const float* W1  = (const float*)d_in[8];
  const float* b1  = (const float*)d_in[9];
  const float* W2  = (const float*)d_in[10];
  const float* b2  = (const float*)d_in[11];
  const float* gam = (const float*)d_in[12];
  const float* bet = (const float*)d_in[13];
  const float* Wh1 = (const float*)d_in[14];
  const float* bh1 = (const float*)d_in[15];
  const float* Wh2 = (const float*)d_in[16];
  const float* bh2 = (const float*)d_in[17];
  float* out = (float*)d_out;

  char* ws = (char*)d_ws;
  auto alloc = [&](size_t bytes) -> char* {
    char* p = ws;
    ws += (bytes + 255) & ~(size_t)255;
    return p;
  };
  const long long NC = (long long)N_NODES * CHP;
  bf16_t* h     = (bf16_t*)alloc(NC * sizeof(bf16_t));
  float*  aggr  = (float*) alloc(NC * sizeof(float));      // doubles as z2
  bf16_t* zin   = (bf16_t*)alloc(NC * sizeof(bf16_t));
  bf16_t* mid   = (bf16_t*)alloc(NC * sizeof(bf16_t));
  bf16_t* Wt    = (bf16_t*)alloc((size_t)NUM_LAYERS * 2 * CHP * CHP * sizeof(bf16_t));
  float*  stats = (float*) alloc(2 * CHP * sizeof(float));
  float*  psums = (float*) alloc((size_t)N_GRAPHS * CHP * sizeof(float));
  float*  pcnts = (float*) alloc(N_GRAPHS * sizeof(float));

  const dim3 blk(256);
  const dim3 gemm_grid((N_NODES + 127) / 128, CHP / 128);
  const long long perW = (long long)CHP * CHP;

  // features
  k_zero_u32<<<4096, blk, 0, stream>>>((uint32_t*)h, NC * 2 / 4);
  k_build_features<<<(N_NODES + 255) / 256, blk, 0, stream>>>(x_type, x_tok, x_small, h);
  // weights -> transposed padded bf16
  k_pack_w<<<8192, blk, 0, stream>>>(W1, W2, Wt);

  for (int l = 0; l < NUM_LAYERS; l++) {
    k_zero_u32<<<8192, blk, 0, stream>>>((uint32_t*)aggr, NC);
    k_edge_scatter<<<N_EDGES, blk, 0, stream>>>(h, edge_idx, edge_typ,
                                                We + (long long)l * NUM_EDGE_TYPES * CH,
                                                be + (long long)l * CH, aggr);
    k_pack_zin<<<8192, blk, 0, stream>>>(h, aggr, zin);
    k_gemm<true,  true ><<<gemm_grid, blk, 0, stream>>>(zin, Wt + (long long)(l * 2 + 0) * perW,
                                                        b1 + (long long)l * CH, mid, nullptr);
    k_gemm<false, false><<<gemm_grid, blk, 0, stream>>>(mid, Wt + (long long)(l * 2 + 1) * perW,
                                                        b2 + (long long)l * CH, nullptr, aggr);
    k_zero_u32<<<10, blk, 0, stream>>>((uint32_t*)stats, 2 * CHP);
    k_bn_stats<<<dim3(CHP / 256, 64), blk, 0, stream>>>(aggr, stats);
    k_bn_finalize<<<CHP / 256, blk, 0, stream>>>(stats);
    k_bn_apply<<<8192, blk, 0, stream>>>(aggr, stats, gam + (long long)l * CH,
                                         bet + (long long)l * CH, h);
  }

  // pool + head
  k_zero_u32<<<256, blk, 0, stream>>>((uint32_t*)psums, (long long)N_GRAPHS * CHP);
  k_zero_u32<<<1, blk, 0, stream>>>((uint32_t*)pcnts, N_GRAPHS);
  k_pool<<<N_NODES, blk, 0, stream>>>(h, batch, psums, pcnts);
  k_head<<<N_GRAPHS, blk, 0, stream>>>(psums, pcnts, Wh1, bh1, Wh2, bh2, out);
}
